// ComplexMDTA_26121991094805
// MI455X (gfx1250) — matrix operand order repaired and checked by reading, not compiled, not run on hardware
//
#include <hip/hip_runtime.h>
#include <hip/hip_bf16.h>

typedef __attribute__((ext_vector_type(16))) __bf16 v16bf;
typedef __attribute__((ext_vector_type(8)))  float  v8f;
typedef __attribute__((ext_vector_type(2)))  float  v2f;
typedef int v4i_vs __attribute__((vector_size(16)));   // matches async-builtin pointee

#define HH 192
#define WW 192
#define HW (192*192)      // 36864
#define CIN 128
#define C3  384
#define NB  4
#define NHEAD 8

#define AS1 __attribute__((address_space(1)))
#define AS3 __attribute__((address_space(3)))

#if defined(__has_builtin)
#if __has_builtin(__builtin_amdgcn_global_load_async_to_lds_b128) && \
    __has_builtin(__builtin_amdgcn_s_wait_asynccnt)
#define USE_ASYNC_LDS 1
#endif
#endif

// ---------------------------------------------------------------------------
// Kernel 1/6: complex 1x1 conv as GEMM, bf16 WMMA 16x16x32, f32 accumulate.
// X: [b, CIN, HW] (r,i), W: [OC, CIN] (r,i), Y: [b, OC, HW] (r,i)
// grid = (HW/16, OC/128, B); block = 256 (8 waves, one 16-row o-tile each)
// X tiles staged in LDS as f32 [channel][pixel]; async global->LDS when the
// toolchain exposes the gfx1250 async builtins.
// ---------------------------------------------------------------------------
__global__ __launch_bounds__(256) void cgemm1x1(
    const float* __restrict__ xr, const float* __restrict__ xi,
    const float* __restrict__ wr, const float* __restrict__ wi,
    float* __restrict__ yr, float* __restrict__ yi, int OC)
{
    const int ptile = blockIdx.x;                 // 16-pixel tile
    const int b     = blockIdx.z;
    const int tid   = threadIdx.x;
    const int wave  = tid >> 5;
    const int lane  = tid & 31;
    const int obase = (blockIdx.y * 8 + wave) * 16;

    __shared__ float lbr[32][16];                 // [channel k][pixel n], f32
    __shared__ float lbi[32][16];

    const float* xrb = xr + (size_t)b * CIN * HW + (size_t)ptile * 16;
    const float* xib = xi + (size_t)b * CIN * HW + (size_t)ptile * 16;

    v8f accrr = {}, accii = {}, accri = {}, accir = {};

    for (int k = 0; k < CIN; k += 32) {
        __syncthreads();                          // protect previous reads
#ifdef USE_ASYNC_LDS
        {   // 256 lanes x 16B = whole 4KB (r+i) tile, straight to LDS
            const int part = tid >> 7;            // 0 = real, 1 = imag
            const int idx  = tid & 127;           // 128 x b128 chunks per part
            const int c    = idx >> 2;            // channel 0..31
            const int ch   = (idx & 3) * 4;       // f32 offset within row
            const float* g = (part ? xib : xrb) + (size_t)(k + c) * HW + ch;
            float* l = (part ? &lbi[0][0] : &lbr[0][0]) + c * 16 + ch;
            __builtin_amdgcn_global_load_async_to_lds_b128(
                (AS1 v4i_vs*)g, (AS3 v4i_vs*)l, 0, 0);
        }
        __builtin_amdgcn_s_wait_asynccnt(0);
        __syncthreads();
#else
        {   // fallback: float2 per thread, contiguous LDS rows
            const int c  = tid >> 3;              // 0..31
            const int px = (tid & 7) * 2;         // 0,2,..,14
            const float2 vr = *(const float2*)(xrb + (size_t)(k + c) * HW + px);
            const float2 vi = *(const float2*)(xib + (size_t)(k + c) * HW + px);
            *(float2*)&lbr[c][px] = vr;
            *(float2*)&lbi[c][px] = vi;
        }
        __syncthreads();
#endif

        // A tile (weights) in exact 16-bit 16x32 WMMA A layout:
        // lanes 0-15: row m=lane, K = {k..k+7, k+16..k+23}
        // lanes 16-31: row m=lane-16, K = {k+8..k+15, k+24..k+31}
        const int row  = obase + (lane & 15);
        const int koff = (lane >> 4) * 8;
        const float* pwr = wr + (size_t)row * CIN + k + koff;
        const float* pwi = wi + (size_t)row * CIN + k + koff;
        v16bf a_r, a_i;
#pragma unroll
        for (int j = 0; j < 8; ++j) {
            a_r[j]     = (__bf16)pwr[j];
            a_r[j + 8] = (__bf16)pwr[j + 16];
            a_i[j]     = (__bf16)pwi[j];
            a_i[j + 8] = (__bf16)pwi[j + 16];
        }

        // B tile from LDS in 16-bit 32x16 WMMA B layout, converting f32->bf16:
        // lane: column n=lane&15; lanes 0-15 hold K {0..7,16..23}, lanes 16-31 K {8..15,24..31}
        const int n  = lane & 15;
        const int kb = (lane >> 4) * 8;
        v16bf b_r, b_i;
#pragma unroll
        for (int j = 0; j < 16; ++j) {
            b_r[j] = (__bf16)lbr[kb + j + (j & 8)][n];
            b_i[j] = (__bf16)lbi[kb + j + (j & 8)][n];
        }

        accrr = __builtin_amdgcn_wmma_f32_16x16x32_bf16(false, a_r, false, b_r, (short)0, accrr, false, false);
        accii = __builtin_amdgcn_wmma_f32_16x16x32_bf16(false, a_i, false, b_i, (short)0, accii, false, false);
        accri = __builtin_amdgcn_wmma_f32_16x16x32_bf16(false, a_r, false, b_i, (short)0, accri, false, false);
        accir = __builtin_amdgcn_wmma_f32_16x16x32_bf16(false, a_i, false, b_r, (short)0, accir, false, false);
    }

    // C/D layout: VGPR r -> row m = r + (lane>=16 ? 8 : 0), col n = lane&15
    const int n  = lane & 15;
    const int mh = (lane >> 4) * 8;
    const size_t base = (size_t)b * OC * HW + (size_t)ptile * 16 + n;
#pragma unroll
    for (int r = 0; r < 8; ++r) {
        const int m = obase + mh + r;
        yr[base + (size_t)m * HW] = accrr[r] - accii[r];
        yi[base + (size_t)m * HW] = accri[r] + accir[r];
    }
}

// ---------------------------------------------------------------------------
// Kernel 2: depthwise 3x3 complex conv (pure VALU, memory streaming)
// ---------------------------------------------------------------------------
__global__ __launch_bounds__(256) void dwconv3x3c(
    const float* __restrict__ yr, const float* __restrict__ yi,
    const float* __restrict__ wr, const float* __restrict__ wi,
    float* __restrict__ zr, float* __restrict__ zi)
{
    size_t idx = (size_t)blockIdx.x * 256 + threadIdx.x;
    int w = (int)(idx % WW);
    size_t t = idx / WW;
    int h = (int)(t % HH);
    size_t bc = t / HH;                   // b*384 + c
    int c = (int)(bc % C3);

    const float* wrp = wr + c * 9;
    const float* wip = wi + c * 9;
    const float* pr  = yr + bc * HW;
    const float* pi  = yi + bc * HW;

    float sr = 0.f, si = 0.f;
#pragma unroll
    for (int dy = -1; dy <= 1; ++dy) {
        int hh = h + dy;
        if (hh < 0 || hh >= HH) continue;
#pragma unroll
        for (int dx = -1; dx <= 1; ++dx) {
            int ww2 = w + dx;
            if (ww2 < 0 || ww2 >= WW) continue;
            float fr = wrp[(dy + 1) * 3 + dx + 1];
            float fi = wip[(dy + 1) * 3 + dx + 1];
            float vr = pr[hh * WW + ww2];
            float vi = pi[hh * WW + ww2];
            sr += fr * vr - fi * vi;
            si += fr * vi + fi * vr;
        }
    }
    zr[idx] = sr;
    zi[idx] = si;
}

// ---------------------------------------------------------------------------
// Kernel 3: inverse L2 norms of q,k rows (channels 0..255 of z), per b
// grid = (256, B), block = 256
// ---------------------------------------------------------------------------
__global__ __launch_bounds__(256) void rownorm(
    const float* __restrict__ zr, const float* __restrict__ zi,
    float* __restrict__ invr, float* __restrict__ invi)
{
    const int c = blockIdx.x;             // 0..255 (q then k chunk)
    const int b = blockIdx.y;
    const float* pr = zr + ((size_t)b * C3 + c) * HW;
    const float* pi = zi + ((size_t)b * C3 + c) * HW;

    float sr = 0.f, si = 0.f;
    for (int t = threadIdx.x; t < HW; t += 256) {
        float a = pr[t]; sr += a * a;
        float d = pi[t]; si += d * d;
    }
    __shared__ float lr[256], li[256];
    lr[threadIdx.x] = sr; li[threadIdx.x] = si;
    __syncthreads();
    for (int s = 128; s > 0; s >>= 1) {
        if (threadIdx.x < s) {
            lr[threadIdx.x] += lr[threadIdx.x + s];
            li[threadIdx.x] += li[threadIdx.x + s];
        }
        __syncthreads();
    }
    if (threadIdx.x == 0) {
        invr[b * 256 + c] = 1.0f / fmaxf(sqrtf(lr[0]), 1e-12f);
        invi[b * 256 + c] = 1.0f / fmaxf(sqrtf(li[0]), 1e-12f);
    }
}

// ---------------------------------------------------------------------------
// Kernel 4: A = softmax(temp * (norm(q) @ norm(k)^T)) complex, per (b,h)
// f32 WMMA 16x16x4 over the long n dimension. grid = 32, block = 256.
// ---------------------------------------------------------------------------
__global__ __launch_bounds__(256) void qkt_softmax(
    const float* __restrict__ zr, const float* __restrict__ zi,
    const float* __restrict__ invr, const float* __restrict__ invi,
    const float* __restrict__ tr, const float* __restrict__ ti,
    float* __restrict__ Ar, float* __restrict__ Ai)
{
    const int bh = blockIdx.x;
    const int b = bh >> 3, h = bh & 7;
    const int wave = threadIdx.x >> 5;
    const int lane = threadIdx.x & 31;

    const size_t zb = (size_t)b * C3 * HW;
    const float* qrp = zr + zb + (size_t)(h * 16) * HW;
    const float* qip = zi + zb + (size_t)(h * 16) * HW;
    const float* krp = zr + zb + (size_t)(128 + h * 16) * HW;
    const float* kip = zi + zb + (size_t)(128 + h * 16) * HW;

    // f32 16x16x4 operand layout: A lane m=lane&15, K = (lane>>4)*2 + {0,1}
    //                             B lane e=lane&15, K = (lane>>4)*2 + {0,1}
    const int m  = lane & 15;
    const int kc = (lane >> 4) * 2;
    const size_t rowm = (size_t)m * HW;

    v8f arr = {}, aii = {}, ari = {}, air = {};
    const int stripe = HW / 8;
    const int n0 = wave * stripe;
    for (int n = n0; n < n0 + stripe; n += 4) {
        v2f aqr = *(const v2f*)(qrp + rowm + n + kc);
        v2f aqi = *(const v2f*)(qip + rowm + n + kc);
        v2f bkr = *(const v2f*)(krp + rowm + n + kc);
        v2f bki = *(const v2f*)(kip + rowm + n + kc);
        arr = __builtin_amdgcn_wmma_f32_16x16x4_f32(false, aqr, false, bkr, (short)0, arr, false, false);
        aii = __builtin_amdgcn_wmma_f32_16x16x4_f32(false, aqi, false, bki, (short)0, aii, false, false);
        ari = __builtin_amdgcn_wmma_f32_16x16x4_f32(false, aqr, false, bki, (short)0, ari, false, false);
        air = __builtin_amdgcn_wmma_f32_16x16x4_f32(false, aqi, false, bkr, (short)0, air, false, false);
    }

    __shared__ float red[8][4][8][32];
#pragma unroll
    for (int j = 0; j < 8; ++j) {
        red[wave][0][j][lane] = arr[j];
        red[wave][1][j][lane] = aii[j];
        red[wave][2][j][lane] = ari[j];
        red[wave][3][j][lane] = air[j];
    }
    __syncthreads();

    if (wave == 0) {
        const int dlo = (lane >> 4) * 8;
        const int e   = lane & 15;
        const float tscr = tr[h], tsci = ti[h];
        float fr[8], fi[8];
#pragma unroll
        for (int j = 0; j < 8; ++j) {
            float s0 = 0, s1 = 0, s2 = 0, s3 = 0;
            for (int w2 = 0; w2 < 8; ++w2) {
                s0 += red[w2][0][j][lane];
                s1 += red[w2][1][j][lane];
                s2 += red[w2][2][j][lane];
                s3 += red[w2][3][j][lane];
            }
            const int d = j + dlo;
            float iqr = invr[b * 256 + h * 16 + d];
            float iqi = invi[b * 256 + h * 16 + d];
            float ikr = invr[b * 256 + 128 + h * 16 + e];
            float iki = invi[b * 256 + 128 + h * 16 + e];
            fr[j] = (iqr * ikr * s0 - iqi * iki * s1) * tscr;
            fi[j] = (iqr * iki * s2 + iqi * ikr * s3) * tsci;
        }
        // softmax over e: 16-wide shuffles inside each wave32 half
#pragma unroll
        for (int j = 0; j < 8; ++j) {
            float mr = fr[j], mi = fi[j];
            for (int o = 8; o > 0; o >>= 1) {
                mr = fmaxf(mr, __shfl_xor(mr, o, 16));
                mi = fmaxf(mi, __shfl_xor(mi, o, 16));
            }
            float er = expf(fr[j] - mr), ei = expf(fi[j] - mi);
            float srr = er, sii = ei;
            for (int o = 8; o > 0; o >>= 1) {
                srr += __shfl_xor(srr, o, 16);
                sii += __shfl_xor(sii, o, 16);
            }
            const int d = j + dlo;
            Ar[(size_t)bh * 256 + d * 16 + e] = er / srr;
            Ai[(size_t)bh * 256 + d * 16 + e] = ei / sii;
        }
    }
}

// ---------------------------------------------------------------------------
// Kernel 5: o = A @ v complex, f32 WMMA 16x16x4 (K=16 in 4 steps)
// grid = (HW/16/8, B*HEADS), block = 256 (8 waves, one pixel tile each)
// ---------------------------------------------------------------------------
__global__ __launch_bounds__(256) void av_gemm(
    const float* __restrict__ zr, const float* __restrict__ zi,
    const float* __restrict__ Ar, const float* __restrict__ Ai,
    float* __restrict__ o_r, float* __restrict__ o_i)
{
    const int bh = blockIdx.y;
    const int b = bh >> 3, h = bh & 7;
    const int wave = threadIdx.x >> 5;
    const int lane = threadIdx.x & 31;
    const int ptile = blockIdx.x * 8 + wave;

    const float* vrp = zr + (size_t)b * C3 * HW + (size_t)(256 + h * 16) * HW;
    const float* vip = zi + (size_t)b * C3 * HW + (size_t)(256 + h * 16) * HW;
    const float* Arp = Ar + (size_t)bh * 256;
    const float* Aip = Ai + (size_t)bh * 256;

    const int m  = lane & 15;
    const int k2 = (lane >> 4) * 2;
    const int pn = ptile * 16 + (lane & 15);

    v8f orr = {}, oii = {}, ori = {}, oir = {};
#pragma unroll
    for (int kk = 0; kk < 4; ++kk) {
        const int e0 = kk * 4 + k2;
        v2f ar_ = *(const v2f*)(Arp + m * 16 + e0);
        v2f ai_ = *(const v2f*)(Aip + m * 16 + e0);
        v2f bvr, bvi;
        bvr[0] = vrp[(size_t)e0 * HW + pn];
        bvr[1] = vrp[(size_t)(e0 + 1) * HW + pn];
        bvi[0] = vip[(size_t)e0 * HW + pn];
        bvi[1] = vip[(size_t)(e0 + 1) * HW + pn];
        orr = __builtin_amdgcn_wmma_f32_16x16x4_f32(false, ar_, false, bvr, (short)0, orr, false, false);
        oii = __builtin_amdgcn_wmma_f32_16x16x4_f32(false, ai_, false, bvi, (short)0, oii, false, false);
        ori = __builtin_amdgcn_wmma_f32_16x16x4_f32(false, ar_, false, bvi, (short)0, ori, false, false);
        oir = __builtin_amdgcn_wmma_f32_16x16x4_f32(false, ai_, false, bvr, (short)0, oir, false, false);
    }

    const int mh = (lane >> 4) * 8;
    const size_t obase = ((size_t)b * CIN + h * 16) * HW + (size_t)ptile * 16 + (lane & 15);
#pragma unroll
    for (int r = 0; r < 8; ++r) {
        const int d = r + mh;
        o_r[obase + (size_t)d * HW] = orr[r] - oii[r];
        o_i[obase + (size_t)d * HW] = ori[r] + oir[r];
    }
}

// ---------------------------------------------------------------------------
extern "C" void kernel_launch(void* const* d_in, const int* in_sizes, int n_in,
                              void* d_out, int out_size, void* d_ws, size_t ws_size,
                              hipStream_t stream)
{
    const float* xr      = (const float*)d_in[0];
    const float* xi      = (const float*)d_in[1];
    const float* qkv_wr  = (const float*)d_in[2];
    const float* qkv_wi  = (const float*)d_in[3];
    const float* dw_wr   = (const float*)d_in[4];
    const float* dw_wi   = (const float*)d_in[5];
    const float* proj_wr = (const float*)d_in[6];
    const float* proj_wi = (const float*)d_in[7];
    const float* temp_r  = (const float*)d_in[8];
    const float* temp_i  = (const float*)d_in[9];

    const size_t S3 = (size_t)NB * C3 * HW;       // 56,623,104 floats
    float* yr   = (float*)d_ws;
    float* yi   = yr + S3;
    float* zr   = yi + S3;
    float* zi   = zr + S3;
    float* invr = zi + S3;                        // 1024
    float* invi = invr + 1024;
    float* Ar   = invi + 1024;                    // 8192
    float* Ai   = Ar + 8192;
    float* o_r  = yr;                             // y is dead after dwconv
    float* o_i  = yi;

    float* out_r = (float*)d_out;
    float* out_i = out_r + (size_t)NB * CIN * HW;

    // 1. qkv complex 1x1 GEMM -> y
    cgemm1x1<<<dim3(HW / 16, 3, NB), 256, 0, stream>>>(xr, xi, qkv_wr, qkv_wi, yr, yi, C3);
    // 2. depthwise 3x3 complex -> z
    dwconv3x3c<<<dim3((unsigned)((size_t)NB * C3 * HW / 256)), 256, 0, stream>>>(yr, yi, dw_wr, dw_wi, zr, zi);
    // 3. inverse L2 norms of q,k rows
    rownorm<<<dim3(256, NB), 256, 0, stream>>>(zr, zi, invr, invi);
    // 4. attention matrices + softmax
    qkt_softmax<<<dim3(NB * NHEAD), 256, 0, stream>>>(zr, zi, invr, invi, temp_r, temp_i, Ar, Ai);
    // 5. A @ v -> o (aliases y)
    av_gemm<<<dim3(HW / 16 / 8, NB * NHEAD), 256, 0, stream>>>(zr, zi, Ar, Ai, o_r, o_i);
    // 6. proj complex 1x1 GEMM -> d_out
    cgemm1x1<<<dim3(HW / 16, 1, NB), 256, 0, stream>>>(o_r, o_i, proj_wr, proj_wi, out_r, out_i, CIN);
}
